// BUTDDecoder_13752485282318
// MI455X (gfx1250) — matrix operand order repaired and checked by reading, not compiled, not run on hardware
//
#include <hip/hip_runtime.h>
#include <hip/hip_fp16.h>

// ---------------- problem constants ----------------
constexpr int kB = 128, kO = 36, kV = 2048, kH = 1024, kE = 1024;
constexpr int kT = 15000, kTp = 15008, kL = 20, kSteps = 19;

typedef __attribute__((ext_vector_type(16))) _Float16 v16h;
typedef __attribute__((ext_vector_type(8)))  _Float16 v8h;
typedef __attribute__((ext_vector_type(8)))  float    v8f;

// ---------------------------------------------------------------------------
// WMMA GEMM: C[M,N] = act(A[M,K] * W[N,K]^T + bias), f16 in / f32 accumulate.
// Block = 8 waves; each wave owns one 16-col N-tile and all 8 M-tiles (128
// rows). A is staged per-block into LDS in 64-wide K chunks (double-buffered,
// 144B padded row stride -> conflict-free fragment reads) and shared by the 8
// waves; weights stream from global/L2 exactly once (grid.y = M/128).
// A fragments are loaded into distinct registers (af[8]) before the WMMA
// burst so ds_loads overlap matrix ops instead of serializing on dscnt==0.
// Requires M%128==0, K%64==0, Nw%16==0. Optionally writes an f16 copy of C.
// ---------------------------------------------------------------------------
__global__ __launch_bounds__(256)
void gemm_f16_wmma(const _Float16* __restrict__ A, int lda,
                   const _Float16* __restrict__ W,          // [Nw, K] row-major
                   const float* __restrict__ bias,          // [Nw]
                   float* __restrict__ C, int ldc,
                   _Float16* __restrict__ Ch, int ldh,      // optional (may be null)
                   int M, int N, int Nw, int K, int act) {
  constexpr int MT = 8;               // M-tiles per wave (128 rows)
  constexpr int CHUNK = 64;           // K elems staged per barrier
  constexpr int LDSROW = 72;          // padded row stride in elems (144 bytes)
  constexpr int BUFSZ = 128 * LDSROW; // 18KB per buffer
  __shared__ __align__(16) _Float16 sm[2 * BUFSZ];

  const int lane = threadIdx.x & 31;
  const int wave = threadIdx.x >> 5;
  const int half = lane >> 4;
  const int m16  = lane & 15;
  const int n0   = (blockIdx.x * 8 + wave) * 16;
  const bool active = (n0 < Nw);
  const int mbase = blockIdx.y * 128;

  // staging: each thread copies 64B (32 f16) of the A chunk per iteration
  const int srow = threadIdx.x >> 1;          // 0..127
  const int scol = (threadIdx.x & 1) * 32;
  const _Float16* sga = A + (size_t)(mbase + srow) * lda + scol;
  _Float16* sla = sm + srow * LDSROW + scol;

  const _Float16* wrow = W + (size_t)((active ? n0 : 0) + m16) * K;

  v8f acc[MT] = {};

  // prime buffer 0 with chunk 0
  {
    v8h d0 = *(const v8h*)(sga);
    v8h d1 = *(const v8h*)(sga + 8);
    v8h d2 = *(const v8h*)(sga + 16);
    v8h d3 = *(const v8h*)(sga + 24);
    *(v8h*)(sla)      = d0;
    *(v8h*)(sla + 8)  = d1;
    *(v8h*)(sla + 16) = d2;
    *(v8h*)(sla + 24) = d3;
  }
  __syncthreads();

  const int nchunks = K / CHUNK;
  for (int c = 0; c < nchunks; ++c) {
    const int cur = c & 1;
    const bool more = (c + 1) < nchunks;
    v8h d0 = {}, d1 = {}, d2 = {}, d3 = {};
    if (more) {                       // global -> regs for next chunk
      const _Float16* g = sga + (c + 1) * CHUNK;
      d0 = *(const v8h*)(g);
      d1 = *(const v8h*)(g + 8);
      d2 = *(const v8h*)(g + 16);
      d3 = *(const v8h*)(g + 24);
    }
    if (active) {
      const int kb = c * CHUNK;
#pragma unroll
      for (int ks = 0; ks < CHUNK; ks += 32) {
        // B fragment: column n0+m16, K = ks+half*8..+7 and ks+16+half*8..+7
        v8h blo = *(const v8h*)(wrow + kb + ks + half * 8);
        v8h bhi = *(const v8h*)(wrow + kb + ks + 16 + half * 8);
        v16h bf = __builtin_shufflevector(blo, bhi,
                    0,1,2,3,4,5,6,7,8,9,10,11,12,13,14,15);
        const _Float16* sbase = sm + cur * BUFSZ + ks + half * 8;
        v16h af[MT];
#pragma unroll
        for (int t = 0; t < MT; ++t) {    // gather all fragments first
          const _Float16* srcrow = sbase + (t * 16 + m16) * LDSROW;
          v8h lo = *(const v8h*)(srcrow);        // K = ks+half*8 .. +7
          v8h hi = *(const v8h*)(srcrow + 16);   // K = ks+16+half*8 .. +7
          af[t] = __builtin_shufflevector(lo, hi,
                      0,1,2,3,4,5,6,7,8,9,10,11,12,13,14,15);
        }
#pragma unroll
        for (int t = 0; t < MT; ++t)      // then the WMMA burst
          acc[t] = __builtin_amdgcn_wmma_f32_16x16x32_f16(
              false, af[t], false, bf, (short)0, acc[t], false, false);
      }
    }
    if (more) {                       // regs -> LDS (other buffer)
      _Float16* l = sla + (cur ^ 1) * BUFSZ;
      *(v8h*)(l)      = d0;
      *(v8h*)(l + 8)  = d1;
      *(v8h*)(l + 16) = d2;
      *(v8h*)(l + 24) = d3;
    }
    __syncthreads();
  }

  const int col = n0 + m16;
  if (active && col < N) {
    const float bv = bias ? bias[col] : 0.0f;
#pragma unroll
    for (int t = 0; t < MT; ++t) {
      const int rbase = mbase + t * 16 + half * 8;
#pragma unroll
      for (int r = 0; r < 8; ++r) {
        const int row = rbase + r;
        if (row < M) {
          float val = acc[t][r] + bv;
          if (act) val = fmaxf(val, 0.0f);
          C[(size_t)row * ldc + col] = val;
          if (Ch) Ch[(size_t)row * ldh + col] = (_Float16)val;
        }
      }
    }
  }
}

// ---------------- helpers ----------------
__global__ void zero_kernel(float* __restrict__ d, size_t n) {
  for (size_t i = (size_t)blockIdx.x * blockDim.x + threadIdx.x; i < n;
       i += (size_t)gridDim.x * blockDim.x)
    d[i] = 0.0f;
}

__global__ void cvt_f16_kernel(const float* __restrict__ s, _Float16* __restrict__ d, size_t n) {
  for (size_t i = (size_t)blockIdx.x * blockDim.x + threadIdx.x; i < n;
       i += (size_t)gridDim.x * blockDim.x)
    d[i] = (_Float16)s[i];
}

__global__ void cvt_pad_rows_kernel(const float* __restrict__ s, _Float16* __restrict__ d,
                                    int rows, int rows_pad, int cols) {
  const size_t n = (size_t)rows_pad * cols;
  for (size_t i = (size_t)blockIdx.x * blockDim.x + threadIdx.x; i < n;
       i += (size_t)gridDim.x * blockDim.x) {
    size_t r = i / cols;
    d[i] = (r < (size_t)rows) ? (_Float16)s[(size_t)r * cols + (i - r * cols)]
                              : (_Float16)0.0f;
  }
}

__global__ void pad_bias_kernel(const float* __restrict__ s, float* __restrict__ d) {
  int i = blockIdx.x * 256 + threadIdx.x;
  if (i < kTp) d[i] = (i < kT) ? s[i] : 0.0f;
}

// Stable descending argsort over cap_len (B=128, single block), plus caption out.
__global__ void sort_kernel(const int* __restrict__ cap_len, const int* __restrict__ caption,
                            int* __restrict__ sort_id, int* __restrict__ dec_len,
                            float* __restrict__ cap_out) {
  __shared__ int len[kB];
  const int i = threadIdx.x;
  len[i] = cap_len[i];
  __syncthreads();
  const int li = len[i];
  int rank = 0;
  for (int j = 0; j < kB; ++j) {
    int lj = len[j];
    if (lj > li || (lj == li && j < i)) rank++;
  }
  sort_id[rank] = i;
  dec_len[rank] = li - 1;
  for (int j = 0; j < kSteps; ++j)
    cap_out[rank * kSteps + j] = (float)caption[i * kL + 1 + j];
}

__global__ void init_state_kernel(float* __restrict__ h1, float* __restrict__ h2,
                                  _Float16* __restrict__ h1h, _Float16* __restrict__ x1) {
  int idx = blockIdx.x * 256 + threadIdx.x;   // kB*kH
  int b = idx >> 10, j = idx & 1023;
  h1[idx] = 0.0f; h2[idx] = 0.0f;
  h1h[idx] = (_Float16)0.0f;
  x1[(size_t)b * 4096 + j] = (_Float16)0.0f;  // h2 slot of x1
}

__global__ void vmean_kernel(const float* __restrict__ v, const int* __restrict__ sid,
                             _Float16* __restrict__ x1) {
  int idx = blockIdx.x * 256 + threadIdx.x;   // kB*kV
  int b = idx >> 11, d = idx & 2047;
  const float* vb = v + (size_t)sid[b] * kO * kV + d;
  float s = 0.0f;
  for (int o = 0; o < kO; ++o) s += vb[(size_t)o * kV];
  x1[(size_t)b * 4096 + 1024 + d] = (_Float16)(s * (1.0f / 36.0f));
}

__global__ void sortv_f16_kernel(const float* __restrict__ v, const int* __restrict__ sid,
                                 _Float16* __restrict__ d) {
  const size_t n = (size_t)kB * kO * kV;
  for (size_t i = (size_t)blockIdx.x * 256 + threadIdx.x; i < n;
       i += (size_t)gridDim.x * 256) {
    int b = (int)(i / (kO * kV));
    size_t rem = i - (size_t)b * (kO * kV);
    d[i] = (_Float16)v[(size_t)sid[b] * (kO * kV) + rem];
  }
}

__global__ void cap_gather_kernel(const float* __restrict__ emb, const int* __restrict__ caption,
                                  const int* __restrict__ sid, _Float16* __restrict__ x1, int t) {
  int idx = blockIdx.x * 256 + threadIdx.x;   // kB*kE
  int b = idx >> 10, e = idx & 1023;
  int tok = caption[sid[b] * kL + t];
  x1[(size_t)b * 4096 + 3072 + e] = (_Float16)emb[(size_t)tok * kE + e];
}

__global__ __launch_bounds__(256)
void gru_cell_kernel(const float* __restrict__ gi, const float* __restrict__ gh,
                     float* __restrict__ h, _Float16* __restrict__ h16, int ld16) {
  int idx = blockIdx.x * 256 + threadIdx.x;   // kB*kH
  int b = idx >> 10, j = idx & 1023;
  const float* gib = gi + (size_t)b * 3072;
  const float* ghb = gh + (size_t)b * 3072;
  float r = 1.0f / (1.0f + __expf(-(gib[j] + ghb[j])));
  float z = 1.0f / (1.0f + __expf(-(gib[1024 + j] + ghb[1024 + j])));
  float n = tanhf(gib[2048 + j] + r * ghb[2048 + j]);
  float hv = (1.0f - z) * n + z * h[idx];
  h[idx] = hv;
  h16[(size_t)b * ld16 + j] = (_Float16)hv;
}

__global__ __launch_bounds__(128)
void att_logits_kernel(const float* __restrict__ vproj, const float* __restrict__ q,
                       const float* __restrict__ Wa, const float* __restrict__ ba,
                       float* __restrict__ logits) {
  int b = blockIdx.y;
  int o = blockIdx.x * 4 + (threadIdx.x >> 5);
  int lane = threadIdx.x & 31;
  if (o >= kO) return;
  const float* vp = vproj + ((size_t)b * kO + o) * kH;
  const float* qb = q + (size_t)b * kH;
  float s = 0.0f;
  for (int j = lane; j < kH; j += 32) s += vp[j] * qb[j] * Wa[j];
  for (int off = 16; off > 0; off >>= 1) s += __shfl_xor(s, off, 32);
  if (lane == 0) logits[b * kO + o] = s + ba[0];
}

__global__ __launch_bounds__(256)
void att_soft_kernel(const float* __restrict__ logits, const float* __restrict__ v,
                     const int* __restrict__ sid, const int* __restrict__ dlen,
                     _Float16* __restrict__ x2, float* __restrict__ alpha_out, int t) {
  const int b = blockIdx.x;
  const int tid = threadIdx.x;
  __shared__ float sm[kO];
  __shared__ float att[kO];
  if (tid < kO) sm[tid] = logits[b * kO + tid];
  __syncthreads();
  float mx = -3.0e38f;
  for (int o = 0; o < kO; ++o) mx = fmaxf(mx, sm[o]);
  float ssum = 0.0f;
  for (int o = 0; o < kO; ++o) ssum += __expf(sm[o] - mx);
  if (tid < kO) {
    float a = __expf(sm[tid] - mx) / ssum;
    att[tid] = a;
    if (t < dlen[b])
      alpha_out[((size_t)sid[b] * kL + t) * kO + tid] = a;
  }
  __syncthreads();
  const float* vb = v + (size_t)sid[b] * kO * kV;
  for (int d = tid; d < kV; d += 256) {
    float s = 0.0f;
    for (int o = 0; o < kO; ++o) s += att[o] * vb[(size_t)o * kV + d];
    x2[(size_t)b * 3072 + d] = (_Float16)s;   // att_v slot of x2
  }
}

__global__ void scatter_out_kernel(const float* __restrict__ tmp, const int* __restrict__ sid,
                                   const int* __restrict__ dlen, float* __restrict__ out, int t) {
  const size_t n = (size_t)kB * kT;
  for (size_t i = (size_t)blockIdx.x * 256 + threadIdx.x; i < n;
       i += (size_t)gridDim.x * 256) {
    int b = (int)(i / kT);
    int c = (int)(i - (size_t)b * kT);
    if (t < dlen[b])
      out[((size_t)sid[b] * kL + t) * kT + c] = tmp[(size_t)b * kTp + c];
  }
}

__global__ __launch_bounds__(256)
void softmax_rows_kernel(float* __restrict__ out) {
  float* row = out + (size_t)blockIdx.x * kT;
  __shared__ float red[256];
  const int tid = threadIdx.x;
  float mx = -3.0e38f;
  for (int i = tid; i < kT; i += 256) mx = fmaxf(mx, row[i]);
  red[tid] = mx; __syncthreads();
  for (int s = 128; s > 0; s >>= 1) { if (tid < s) red[tid] = fmaxf(red[tid], red[tid + s]); __syncthreads(); }
  mx = red[0]; __syncthreads();
  float sum = 0.0f;
  for (int i = tid; i < kT; i += 256) sum += __expf(row[i] - mx);
  red[tid] = sum; __syncthreads();
  for (int s = 128; s > 0; s >>= 1) { if (tid < s) red[tid] += red[tid + s]; __syncthreads(); }
  const float inv = 1.0f / red[0];
  __syncthreads();
  for (int i = tid; i < kT; i += 256) row[i] = __expf(row[i] - mx) * inv;
}

// ---------------- host ----------------
static inline void launch_gemm(hipStream_t s, const _Float16* A, int lda, const _Float16* W,
                               const float* bias, float* C, int ldc, _Float16* Ch, int ldh,
                               int M, int N, int Nw, int K, int act) {
  dim3 g((Nw / 16 + 7) / 8, M / 128);
  gemm_f16_wmma<<<g, 256, 0, s>>>(A, lda, W, bias, C, ldc, Ch, ldh, M, N, Nw, K, act);
}

extern "C" void kernel_launch(void* const* d_in, const int* in_sizes, int n_in,
                              void* d_out, int out_size, void* d_ws, size_t ws_size,
                              hipStream_t stream) {
  const float* v       = (const float*)d_in[0];
  const int*   caption = (const int*)d_in[1];
  const int*   cap_len = (const int*)d_in[2];
  const float* emb     = (const float*)d_in[3];
  const float* Wih1    = (const float*)d_in[4];
  const float* Whh1    = (const float*)d_in[5];
  const float* bih1    = (const float*)d_in[6];
  const float* bhh1    = (const float*)d_in[7];
  const float* Wih2    = (const float*)d_in[8];
  const float* Whh2    = (const float*)d_in[9];
  const float* bih2    = (const float*)d_in[10];
  const float* bhh2    = (const float*)d_in[11];
  const float* Wfc1    = (const float*)d_in[12];
  const float* bfc1    = (const float*)d_in[13];
  const float* Wfc2    = (const float*)d_in[14];
  const float* bfc2    = (const float*)d_in[15];
  const float* Wv      = (const float*)d_in[16];
  const float* bv      = (const float*)d_in[17];
  const float* Wq      = (const float*)d_in[18];
  const float* bq      = (const float*)d_in[19];
  const float* Wa      = (const float*)d_in[20];
  const float* ba      = (const float*)d_in[21];

  float* out_predict = (float*)d_out;                              // [B,20,15000]
  float* out_caption = out_predict + (size_t)kB * kL * kT;         // [B,19]
  float* out_alpha   = out_caption + (size_t)kB * kSteps;          // [B,20,36]

  // ---- workspace carve-out ----
  char* wp = (char*)d_ws;
  auto alloc = [&](size_t bytes) -> void* {
    void* r = (void*)wp;
    wp += (bytes + 255) & ~(size_t)255;
    return r;
  };
  _Float16* Wih1h = (_Float16*)alloc((size_t)3072 * 4096 * 2);
  _Float16* Whh1h = (_Float16*)alloc((size_t)3072 * 1024 * 2);
  _Float16* Wih2h = (_Float16*)alloc((size_t)3072 * 3072 * 2);
  _Float16* Whh2h = (_Float16*)alloc((size_t)3072 * 1024 * 2);
  _Float16* Wfc1h = (_Float16*)alloc((size_t)1024 * 1024 * 2);
  _Float16* Wqh   = (_Float16*)alloc((size_t)1024 * 1024 * 2);
  _Float16* Wvh   = (_Float16*)alloc((size_t)1024 * 2048 * 2);
  _Float16* Wfc2h = (_Float16*)alloc((size_t)kTp * 1024 * 2);
  float*    bfc2p = (float*)alloc((size_t)kTp * 4);
  _Float16* v16   = (_Float16*)alloc((size_t)kB * kO * kV * 2);
  float*    vproj = (float*)alloc((size_t)kB * kO * kH * 4);
  _Float16* x1    = (_Float16*)alloc((size_t)kB * 4096 * 2);  // [h2 | v_mean | cap_t]
  _Float16* x2    = (_Float16*)alloc((size_t)kB * 3072 * 2);  // [att_v | h]
  float*    h1    = (float*)alloc((size_t)kB * kH * 4);
  float*    h2    = (float*)alloc((size_t)kB * kH * 4);
  _Float16* h1h   = (_Float16*)alloc((size_t)kB * kH * 2);
  float*    hbuf  = (float*)alloc((size_t)kB * kH * 4);
  float*    q     = (float*)alloc((size_t)kB * kH * 4);
  float*    gi1   = (float*)alloc((size_t)kB * 3072 * 4);
  float*    gh1   = (float*)alloc((size_t)kB * 3072 * 4);
  float*    gi2   = (float*)alloc((size_t)kB * 3072 * 4);
  float*    gh2   = (float*)alloc((size_t)kB * 3072 * 4);
  float*    logits= (float*)alloc((size_t)kB * kO * 4);
  float*    otmp  = (float*)alloc((size_t)kB * kTp * 4);
  int*      sid   = (int*)alloc(kB * 4);
  int*      dlen  = (int*)alloc(kB * 4);

  // ---- setup ----
  zero_kernel<<<4096, 256, 0, stream>>>((float*)d_out, (size_t)out_size);
  cvt_f16_kernel<<<4096, 256, 0, stream>>>(Wih1, Wih1h, (size_t)3072 * 4096);
  cvt_f16_kernel<<<4096, 256, 0, stream>>>(Whh1, Whh1h, (size_t)3072 * 1024);
  cvt_f16_kernel<<<4096, 256, 0, stream>>>(Wih2, Wih2h, (size_t)3072 * 3072);
  cvt_f16_kernel<<<4096, 256, 0, stream>>>(Whh2, Whh2h, (size_t)3072 * 1024);
  cvt_f16_kernel<<<4096, 256, 0, stream>>>(Wfc1, Wfc1h, (size_t)1024 * 1024);
  cvt_f16_kernel<<<4096, 256, 0, stream>>>(Wq,   Wqh,   (size_t)1024 * 1024);
  cvt_f16_kernel<<<4096, 256, 0, stream>>>(Wv,   Wvh,   (size_t)1024 * 2048);
  cvt_pad_rows_kernel<<<4096, 256, 0, stream>>>(Wfc2, Wfc2h, kT, kTp, 1024);
  pad_bias_kernel<<<(kTp + 255) / 256, 256, 0, stream>>>(bfc2, bfc2p);

  sort_kernel<<<1, kB, 0, stream>>>(cap_len, caption, sid, dlen, out_caption);
  init_state_kernel<<<kB * kH / 256, 256, 0, stream>>>(h1, h2, h1h, x1);
  vmean_kernel<<<kB * kV / 256, 256, 0, stream>>>(v, sid, x1);
  sortv_f16_kernel<<<4096, 256, 0, stream>>>(v, sid, v16);

  // v_proj = relu(v_s @ Wv^T + bv): [4608,1024]
  launch_gemm(stream, v16, kV, Wvh, bv, vproj, kH, nullptr, 0, kB * kO, kH, kH, kV, 1);

  for (int t = 0; t < kSteps; ++t) {
    cap_gather_kernel<<<kB * kE / 256, 256, 0, stream>>>(emb, caption, sid, x1, t);
    // GRU1: gi1 = x1 @ Wih1^T + bih1 ; gh1 = h1 @ Whh1^T + bhh1
    launch_gemm(stream, x1, 4096, Wih1h, bih1, gi1, 3072, nullptr, 0, kB, 3072, 3072, 4096, 0);
    launch_gemm(stream, h1h, 1024, Whh1h, bhh1, gh1, 3072, nullptr, 0, kB, 3072, 3072, 1024, 0);
    gru_cell_kernel<<<kB * kH / 256, 256, 0, stream>>>(gi1, gh1, h1, h1h, 1024);
    // h = h1 @ Wfc1^T + bfc1 (f16 copy -> x2[:,2048:3072])
    launch_gemm(stream, h1h, 1024, Wfc1h, bfc1, hbuf, 1024, x2 + 2048, 3072, kB, 1024, 1024, 1024, 0);
    // q = relu(h @ Wq^T + bq)
    launch_gemm(stream, x2 + 2048, 3072, Wqh, bq, q, 1024, nullptr, 0, kB, 1024, 1024, 1024, 1);
    // attention
    att_logits_kernel<<<dim3(9, kB), 128, 0, stream>>>(vproj, q, Wa, ba, logits);
    att_soft_kernel<<<kB, 256, 0, stream>>>(logits, v, sid, dlen, x2, out_alpha, t);
    // GRU2: gi2 = [att_v,h] @ Wih2^T ; gh2 = h2 @ Whh2^T  (h2 f16 lives in x1[:,0:1024])
    launch_gemm(stream, x2, 3072, Wih2h, bih2, gi2, 3072, nullptr, 0, kB, 3072, 3072, 3072, 0);
    launch_gemm(stream, x1, 4096, Whh2h, bhh2, gh2, 3072, nullptr, 0, kB, 3072, 3072, 1024, 0);
    gru_cell_kernel<<<kB * kH / 256, 256, 0, stream>>>(gi2, gh2, h2, x1, 4096);
    // out_t = h2 @ Wfc2^T + bfc2  (h2 f16 read back from x1[:,0:1024])
    launch_gemm(stream, x1, 4096, Wfc2h, bfc2p, otmp, kTp, nullptr, 0, kB, kT, kTp, 1024, 0);
    scatter_out_kernel<<<4096, 256, 0, stream>>>(otmp, sid, dlen, out_predict, t);
  }

  softmax_rows_kernel<<<kB * kL, 256, 0, stream>>>(out_predict);
}
